// DeepGraphTransformer_mse_75462575391238
// MI455X (gfx1250) — matrix operand order repaired and checked by reading, not compiled, not run on hardware
//
#include <hip/hip_runtime.h>
#include <hip/hip_bf16.h>
#include <math.h>

// ---------------- CDNA5 WMMA types ----------------
typedef __attribute__((ext_vector_type(16))) __bf16 v16bf;
typedef __attribute__((ext_vector_type(8)))  float  v8f;
typedef unsigned short u16;

#define NN    50000
#define EE    100000
#define DIM   256
#define INNER 256
#define HEADS 4
#define DH    64
#define EDIM  512
#define FFD   1024
#define HALF  25000

__device__ __forceinline__ u16 f2bf(float f) {
    unsigned u = __float_as_uint(f);
    unsigned r = u + 0x7FFFu + ((u >> 16) & 1u);   // round-to-nearest-even
    return (u16)(r >> 16);
}

__device__ __forceinline__ float gelu_exact(float x) {
    return 0.5f * x * (1.0f + erff(x * 0.70710678118654752f));
}

// block-wide sum over 256 threads
__device__ __forceinline__ float block_sum_256(float v, float* sh) {
    int tid = threadIdx.x;
    sh[tid] = v;
    __syncthreads();
    for (int s = 128; s > 0; s >>= 1) {
        if (tid < s) sh[tid] += sh[tid + s];
        __syncthreads();
    }
    float r = sh[0];
    __syncthreads();
    return r;
}

// ---------------- weight pack: fp32 KxN row-major -> bf16 WMMA B-fragment order ----------------
// fragment (kt,nt): lane l holds col n = nt*16 + (l&15), elements i=0..15 -> k = kt*32 + (l>>4)*16 + i
__global__ void k_pack_b(const float* __restrict__ W, u16* __restrict__ P, int K, int N) {
    int idx = blockIdx.x * blockDim.x + threadIdx.x;
    int total = K * N;
    if (idx >= total) return;
    int i    = idx & 15;
    int lane = (idx >> 4) & 31;
    int t    = idx >> 9;
    int Ntiles = N >> 4;
    int nt = t % Ntiles;
    int kt = t / Ntiles;
    int k = kt * 32 + (lane >> 4) * 16 + i;
    int n = nt * 16 + (lane & 15);
    P[idx] = f2bf(W[(size_t)k * N + n]);
}

// ---------------- bf16 WMMA GEMM, 4 N-tiles per wave ----------------
// C[M,N] = A[M,K] @ B[K,N] + bias, optional GELU, fp32 or bf16 out.
// One wave computes a 16x64 output strip: one A-fragment load feeds 4 WMMAs.
template <int ACT_GELU, int OUT_BF16>
__global__ __launch_bounds__(256) void k_gemm_bf16(
    const u16* __restrict__ A, const u16* __restrict__ Bp,
    const float* __restrict__ bias, void* __restrict__ Cout,
    int M, int N, int K)
{
    int wave = (int)((blockIdx.x * blockDim.x + threadIdx.x) >> 5);
    int lane = threadIdx.x & 31;
    int Mtiles = M >> 4, Ntiles = N >> 4;
    int Ngroups = Ntiles >> 2;                 // groups of 4 N-tiles
    if (wave >= Mtiles * Ngroups) return;
    int ng = wave % Ngroups;                   // N-group index
    int mt = wave / Ngroups;
    int nt0 = ng * 4;

    int r = lane & 15;       // row-in-tile (A) / col-in-tile (B,C)
    int g = lane >> 4;       // half-wave group

    const u16* arow = A + (size_t)(mt * 16 + r) * K;
    const v16bf* bfrag = (const v16bf*)Bp;

    v8f acc0 = {}, acc1 = {}, acc2 = {}, acc3 = {};
    int Ktiles = K >> 5;
    for (int kt = 0; kt < Ktiles; ++kt) {
        // A fragment: contiguous k order (matches k_pack_b):
        //   group 0: s[0..15]=K 0..15
        //   group 1: s[0..15]=K16..31
        union { v16bf v; u16 s[16]; float4 f4[2]; } a;
        a.f4[0] = *(const float4*)(arow + kt * 32 + g * 16);
        a.f4[1] = *(const float4*)(arow + kt * 32 + g * 16 + 8);
        const v16bf* bk = bfrag + ((size_t)kt * Ntiles + nt0) * 32 + lane;
        v16bf b0 = bk[0];
        v16bf b1 = bk[32];
        v16bf b2 = bk[64];
        v16bf b3 = bk[96];
        acc0 = __builtin_amdgcn_wmma_f32_16x16x32_bf16(false, a.v, false, b0, (short)0, acc0, false, false);
        acc1 = __builtin_amdgcn_wmma_f32_16x16x32_bf16(false, a.v, false, b1, (short)0, acc1, false, false);
        acc2 = __builtin_amdgcn_wmma_f32_16x16x32_bf16(false, a.v, false, b2, (short)0, acc2, false, false);
        acc3 = __builtin_amdgcn_wmma_f32_16x16x32_bf16(false, a.v, false, b3, (short)0, acc3, false, false);
    }

    // store: C layout per tile: VGPR j -> M=j (lanes 0-15) / M=8+j (lanes 16-31), N=lane%16
    v8f accs[4] = {acc0, acc1, acc2, acc3};
#pragma unroll
    for (int t = 0; t < 4; ++t) {
        int col = (nt0 + t) * 16 + r;
        float bv = bias ? bias[col] : 0.0f;
#pragma unroll
        for (int j = 0; j < 8; ++j) {
            int orow = mt * 16 + j + 8 * g;
            float v = accs[t][j] + bv;
            if (ACT_GELU) v = gelu_exact(v);
            if (OUT_BF16) ((u16*)Cout)[(size_t)orow * N + col] = f2bf(v);
            else          ((float*)Cout)[(size_t)orow * N + col] = v;
        }
    }
}

// ---------------- edge_attr = concat(x[src], x[dst]) -> bf16, built once from original x ----------------
__global__ void k_edge_attr(const float* __restrict__ x, const int* __restrict__ src,
                            const int* __restrict__ dst, u16* __restrict__ ea)
{
    size_t idx = (size_t)blockIdx.x * blockDim.x + threadIdx.x;
    if (idx >= (size_t)EE * EDIM) return;
    int e = (int)(idx >> 9);
    int c = (int)(idx & 511);
    float v = (c < DIM) ? x[(size_t)src[e] * DIM + c]
                        : x[(size_t)dst[e] * DIM + (c - DIM)];
    ea[idx] = f2bf(v);
}

// ---------------- LayerNorm row kernel (fp32 in -> bf16 out), 256 threads/row ----------------
__global__ void k_layernorm(const float* __restrict__ X, const float* __restrict__ gamma,
                            const float* __restrict__ beta, u16* __restrict__ Y)
{
    __shared__ float sh[256];
    int row = blockIdx.x, tid = threadIdx.x;
    float v = X[(size_t)row * DIM + tid];
    float mu = block_sum_256(v, sh) * (1.0f / DIM);
    float d = v - mu;
    float var = block_sum_256(d * d, sh) * (1.0f / DIM);
    float xn = d * rsqrtf(var + 1e-5f) * gamma[tid] + beta[tid];
    Y[(size_t)row * DIM + tid] = f2bf(xn);
}

// ---------------- attention scalar pipeline ----------------
__global__ void k_init_attn(float* __restrict__ smax, float* __restrict__ denom,
                            float* __restrict__ agg)
{
    size_t idx = (size_t)blockIdx.x * blockDim.x + threadIdx.x;
    if (idx < (size_t)NN * DIM) agg[idx] = 0.0f;
    if (idx < (size_t)NN * HEADS) { smax[idx] = -INFINITY; denom[idx] = 0.0f; }
}

__global__ void k_sim(const float* __restrict__ q, const float* __restrict__ kv,
                      const float* __restrict__ ebuf, const int* __restrict__ src,
                      const int* __restrict__ dst, float* __restrict__ sim)
{
    int idx = blockIdx.x * blockDim.x + threadIdx.x;
    if (idx >= EE * HEADS) return;
    int e = idx >> 2, h = idx & 3;
    const float* qp = q    + (size_t)dst[e] * INNER + h * DH;
    const float* kp = kv   + (size_t)src[e] * (2 * INNER) + h * DH;   // k = first half of kv
    const float* ep = ebuf + (size_t)e * INNER + h * DH;
    float acc = 0.0f;
#pragma unroll 8
    for (int i = 0; i < DH; ++i) acc += qp[i] * (kp[i] + ep[i]);
    sim[idx] = acc * 0.125f;   // DH^-0.5
}

__device__ __forceinline__ void atomic_max_f32(float* addr, float val) {
    unsigned* ua = (unsigned*)addr;
    unsigned old = *ua;
    while (__uint_as_float(old) < val) {
        unsigned assumed = old;
        old = atomicCAS(ua, assumed, __float_as_uint(val));
        if (old == assumed) break;
    }
}

__global__ void k_seg_max(const float* __restrict__ sim, const int* __restrict__ dst,
                          float* __restrict__ smax)
{
    int idx = blockIdx.x * blockDim.x + threadIdx.x;
    if (idx >= EE * HEADS) return;
    int e = idx >> 2, h = idx & 3;
    atomic_max_f32(&smax[dst[e] * HEADS + h], sim[idx]);
}

__global__ void k_exp_denom(float* __restrict__ sim, const int* __restrict__ dst,
                            const float* __restrict__ smax, float* __restrict__ denom)
{
    int idx = blockIdx.x * blockDim.x + threadIdx.x;
    if (idx >= EE * HEADS) return;
    int e = idx >> 2, h = idx & 3;
    float ez = __expf(sim[idx] - smax[dst[e] * HEADS + h]);
    sim[idx] = ez;
    atomicAdd(&denom[dst[e] * HEADS + h], ez);
}

__global__ void k_aggregate(const float* __restrict__ ez, const float* __restrict__ denom,
                            const float* __restrict__ kv, const float* __restrict__ ebuf,
                            const int* __restrict__ src, const int* __restrict__ dst,
                            float* __restrict__ agg)
{
    int e = blockIdx.x, c = threadIdx.x, h = c >> 6;
    int s = src[e], d = dst[e];
    float attn = ez[e * HEADS + h] / denom[d * HEADS + h];
    float ve = kv[(size_t)s * (2 * INNER) + INNER + c] + ebuf[(size_t)e * INNER + c];
    atomicAdd(&agg[(size_t)d * DIM + c], attn * ve);
}

__global__ void k_f32_to_bf16(const float* __restrict__ in, u16* __restrict__ out, size_t n) {
    size_t idx = (size_t)blockIdx.x * blockDim.x + threadIdx.x;
    if (idx < n) out[idx] = f2bf(in[idx]);
}

// ---------------- gated residual: gate = sigmoid([x,res,x-res]@w); nodes = x*g + res*(1-g) ----------------
__global__ void k_gated_residual(const float* __restrict__ Xa, float* __restrict__ nodes,
                                 const float* __restrict__ gw)
{
    __shared__ float sh[256];
    int row = blockIdx.x, tid = threadIdx.x;
    float xv = Xa[(size_t)row * DIM + tid];
    float rv = nodes[(size_t)row * DIM + tid];
    float p = xv * gw[tid] + rv * gw[DIM + tid] + (xv - rv) * gw[2 * DIM + tid];
    float s = block_sum_256(p, sh);
    float gate = 1.0f / (1.0f + __expf(-s));
    nodes[(size_t)row * DIM + tid] = xv * gate + rv * (1.0f - gate);
}

// ---------------- final projection + split/concat output ----------------
// out = [ z_new(25000) | z_old(25000) | z_new(25000) ]
__global__ void k_final(const float* __restrict__ nodes, const float* __restrict__ Wout,
                        const float* __restrict__ bout, float* __restrict__ out)
{
    __shared__ float sh[256];
    int row = blockIdx.x, tid = threadIdx.x;
    float p = nodes[(size_t)row * DIM + tid] * Wout[tid];
    float s = block_sum_256(p, sh);
    if (tid == 0) {
        float z = s + bout[0];
        if (row < HALF) {
            out[HALF + row] = z;                       // z_old block
        } else {
            out[row - HALF] = z;                       // z_new (first copy)
            out[2 * HALF + (row - HALF)] = z;          // z_new (second copy)
        }
    }
}

// ---------------- host orchestration ----------------
static inline int gemm_grid(int M, int N) {
    int waves = (M >> 4) * ((N >> 4) >> 2);    // 4 N-tiles per wave
    return (waves + 7) / 8;                    // 8 waves (wave32) per 256-thread block
}

extern "C" void kernel_launch(void* const* d_in, const int* in_sizes, int n_in,
                              void* d_out, int out_size, void* d_ws, size_t ws_size,
                              hipStream_t stream)
{
    (void)in_sizes; (void)n_in; (void)out_size; (void)ws_size;

    const float* x      = (const float*)d_in[0];
    const int*   ei     = (const int*)d_in[1];
    const float* ln1_g  = (const float*)d_in[2];
    const float* ln1_b  = (const float*)d_in[3];
    const float* Wq     = (const float*)d_in[4];
    const float* bq     = (const float*)d_in[5];
    const float* Wkv    = (const float*)d_in[6];
    const float* bkv    = (const float*)d_in[7];
    const float* We     = (const float*)d_in[8];
    const float* be     = (const float*)d_in[9];
    const float* Wo     = (const float*)d_in[10];
    const float* bo     = (const float*)d_in[11];
    const float* g_attn = (const float*)d_in[12];
    const float* ln2_g  = (const float*)d_in[13];
    const float* ln2_b  = (const float*)d_in[14];
    const float* W1     = (const float*)d_in[15];
    const float* b1     = (const float*)d_in[16];
    const float* W2     = (const float*)d_in[17];
    const float* b2     = (const float*)d_in[18];
    const float* g_ff   = (const float*)d_in[19];
    const float* Wout   = (const float*)d_in[20];
    const float* bout   = (const float*)d_in[21];

    const int* src = ei;
    const int* dst = ei + EE;
    float* out = (float*)d_out;

    // ---- workspace bump allocator (256B aligned) ----
    char* base = (char*)d_ws;
    size_t off = 0;
    auto alloc = [&](size_t bytes) -> char* {
        char* p = base + off;
        off += (bytes + 255) & ~(size_t)255;
        return p;
    };
    float* nodes    = (float*)alloc((size_t)NN * DIM * 4);
    u16*   ea_bf    = (u16*)  alloc((size_t)EE * EDIM * 2);
    u16*   xn_bf    = (u16*)  alloc((size_t)NN * DIM * 2);
    float* qbuf     = (float*)alloc((size_t)NN * INNER * 4);      // aliased as agg after k_sim
    float* kvbuf    = (float*)alloc((size_t)NN * 2 * INNER * 4);
    float* ebuf     = (float*)alloc((size_t)EE * INNER * 4);
    float* simbuf   = (float*)alloc((size_t)EE * HEADS * 4);
    float* smax     = (float*)alloc((size_t)NN * HEADS * 4);
    float* denom    = (float*)alloc((size_t)NN * HEADS * 4);
    u16*   agg_bf   = (u16*)  alloc((size_t)NN * DIM * 2);
    float* attn_out = (float*)alloc((size_t)NN * DIM * 4);        // aliased as ffout
    u16*   h_bf     = (u16*)  alloc((size_t)NN * FFD * 2);
    u16*   Wq_p     = (u16*)  alloc((size_t)DIM * INNER * 2);
    u16*   Wkv_p    = (u16*)  alloc((size_t)DIM * 2 * INNER * 2);
    u16*   We_p     = (u16*)  alloc((size_t)EDIM * INNER * 2);
    u16*   Wo_p     = (u16*)  alloc((size_t)INNER * DIM * 2);
    u16*   W1_p     = (u16*)  alloc((size_t)DIM * FFD * 2);
    u16*   W2_p     = (u16*)  alloc((size_t)FFD * DIM * 2);
    float* agg      = qbuf;        // reuse (q is dead after k_sim)
    float* ffout    = attn_out;    // reuse (attn_out is dead after first gated residual)

    // ---- residual init + edge features (from original x) ----
    hipMemcpyAsync(nodes, x, (size_t)NN * DIM * 4, hipMemcpyDeviceToDevice, stream);
    {
        size_t tot = (size_t)EE * EDIM;
        k_edge_attr<<<(int)((tot + 255) / 256), 256, 0, stream>>>(x, src, dst, ea_bf);
    }

    const int EH_BLOCKS = (EE * HEADS + 255) / 256;

    for (int l = 0; l < 2; ++l) {
        // pack this layer's weights into WMMA B-fragment order (bf16)
        k_pack_b<<<(DIM * INNER + 255) / 256, 256, 0, stream>>>(Wq + (size_t)l * DIM * INNER, Wq_p, DIM, INNER);
        k_pack_b<<<(DIM * 2 * INNER + 255) / 256, 256, 0, stream>>>(Wkv + (size_t)l * DIM * 2 * INNER, Wkv_p, DIM, 2 * INNER);
        k_pack_b<<<(EDIM * INNER + 255) / 256, 256, 0, stream>>>(We + (size_t)l * EDIM * INNER, We_p, EDIM, INNER);
        k_pack_b<<<(INNER * DIM + 255) / 256, 256, 0, stream>>>(Wo + (size_t)l * INNER * DIM, Wo_p, INNER, DIM);
        k_pack_b<<<(DIM * FFD + 255) / 256, 256, 0, stream>>>(W1 + (size_t)l * DIM * FFD, W1_p, DIM, FFD);
        k_pack_b<<<(FFD * DIM + 255) / 256, 256, 0, stream>>>(W2 + (size_t)l * FFD * DIM, W2_p, FFD, DIM);

        // LN1 -> xn (bf16)
        k_layernorm<<<NN, 256, 0, stream>>>(nodes, ln1_g + l * DIM, ln1_b + l * DIM, xn_bf);

        // q = xn@Wq + bq ; kv = xn@Wkv + bkv ; e = edge_attr@We + be   (WMMA GEMMs)
        k_gemm_bf16<0, 0><<<gemm_grid(NN, INNER), 256, 0, stream>>>(xn_bf, Wq_p, bq + l * INNER, qbuf, NN, INNER, DIM);
        k_gemm_bf16<0, 0><<<gemm_grid(NN, 2 * INNER), 256, 0, stream>>>(xn_bf, Wkv_p, bkv + l * 2 * INNER, kvbuf, NN, 2 * INNER, DIM);
        k_gemm_bf16<0, 0><<<gemm_grid(EE, INNER), 256, 0, stream>>>(ea_bf, We_p, be + l * INNER, ebuf, EE, INNER, EDIM);

        // edge attention (segment softmax over dst)
        k_sim<<<EH_BLOCKS, 256, 0, stream>>>(qbuf, kvbuf, ebuf, src, dst, simbuf);
        {
            size_t tot = (size_t)NN * DIM;
            k_init_attn<<<(int)((tot + 255) / 256), 256, 0, stream>>>(smax, denom, agg);
        }
        k_seg_max<<<EH_BLOCKS, 256, 0, stream>>>(simbuf, dst, smax);
        k_exp_denom<<<EH_BLOCKS, 256, 0, stream>>>(simbuf, dst, smax, denom);
        k_aggregate<<<EE, 256, 0, stream>>>(simbuf, denom, kvbuf, ebuf, src, dst, agg);

        // attn_out = agg @ Wo + bo
        {
            size_t tot = (size_t)NN * DIM;
            k_f32_to_bf16<<<(int)((tot + 255) / 256), 256, 0, stream>>>(agg, agg_bf, tot);
        }
        k_gemm_bf16<0, 0><<<gemm_grid(NN, DIM), 256, 0, stream>>>(agg_bf, Wo_p, bo + l * DIM, attn_out, NN, DIM, INNER);

        // nodes = gated_residual(attn_out, nodes, g_attn[l])
        k_gated_residual<<<NN, 256, 0, stream>>>(attn_out, nodes, g_attn + (size_t)l * 3 * DIM);

        // FF: h = gelu(xn2@W1 + b1) (bf16 out, fused) ; ffout = h@W2 + b2
        k_layernorm<<<NN, 256, 0, stream>>>(nodes, ln2_g + l * DIM, ln2_b + l * DIM, xn_bf);
        k_gemm_bf16<1, 1><<<gemm_grid(NN, FFD), 256, 0, stream>>>(xn_bf, W1_p, b1 + l * FFD, h_bf, NN, FFD, DIM);
        k_gemm_bf16<0, 0><<<gemm_grid(NN, DIM), 256, 0, stream>>>(h_bf, W2_p, b2 + l * DIM, ffout, NN, DIM, FFD);

        // nodes = gated_residual(ffout, nodes, g_ff[l])
        k_gated_residual<<<NN, 256, 0, stream>>>(ffout, nodes, g_ff + (size_t)l * 3 * DIM);
    }

    // z = nodes @ Wout + bout ; write [z_new | z_old | z_new]
    k_final<<<NN, 256, 0, stream>>>(nodes, Wout, bout, out);
}
